// StaticFP8Linear_3109556322454
// MI455X (gfx1250) — 8-bit matrix operand order repaired and checked by reading, compiled, not run on hardware
//
#include <hip/hip_runtime.h>
#include <hip/hip_bf16.h>


typedef int   v16i __attribute__((ext_vector_type(16)));
typedef float v8f  __attribute__((ext_vector_type(8)));
typedef int   v4i_vs __attribute__((vector_size(16)));

#define M_TOTAL 16384
#define N_TOTAL 2048
#define K_TOTAL 2048
#define BM 128
#define BN 128
#define BK 128
#define KSTAGES (K_TOTAL / BK)

#if __has_builtin(__builtin_amdgcn_global_load_async_to_lds_b128)
#define HAVE_ASYNC_LDS 1
#else
#define HAVE_ASYNC_LDS 0
#endif

#if HAVE_ASYNC_LDS
__device__ inline void async_copy_b128(const unsigned char* g, unsigned char* l) {
    __builtin_amdgcn_global_load_async_to_lds_b128(
        (__attribute__((address_space(1))) v4i_vs*)g,
        (__attribute__((address_space(3))) v4i_vs*)l,
         0,  0);
}
#endif

__device__ inline void wait_async0() {
#if __has_builtin(__builtin_amdgcn_s_wait_asynccnt)
    __builtin_amdgcn_s_wait_asynccnt(0);
#else
    asm volatile("s_wait_asynccnt 0" ::: "memory");
#endif
}

__device__ inline unsigned char f32_to_e4m3_sw(float f) {
    unsigned int u   = __float_as_uint(f);
    unsigned int sgn = (u >> 24) & 0x80u;
    unsigned int a   = u & 0x7FFFFFFFu;
    if (a > 0x7F800000u) return (unsigned char)(sgn | 0x7Fu);
    if (a >= 0x43E00000u) return (unsigned char)(sgn | 0x7Eu);
    if (a < 0x3A800000u) {
        float q = __uint_as_float(a) * 512.0f;
        int   r = (int)(q + 0.5f);
        return (unsigned char)(sgn | (unsigned)(r & 7));
    }
    unsigned int lsb = (a >> 20) & 1u;
    a += 0x0007FFFFu + lsb;
    int          e   = (int)((a >> 23) & 0xFFu) - 127;
    unsigned int man = (a >> 20) & 0x7u;
    if (e < -6) {
        int shift = -6 - e;
        man = (man | 8u) >> shift;
        return (unsigned char)(sgn | man);
    }
    return (unsigned char)(sgn | ((unsigned)(e + 7) << 3) | man);
}

__device__ inline unsigned int pack2_fp8(float a, float b) {
#if __has_builtin(__builtin_amdgcn_cvt_pk_fp8_f32)
    return ((unsigned int)__builtin_amdgcn_cvt_pk_fp8_f32(a, b, 0, false)) & 0xFFFFu;
#else
    return (unsigned int)f32_to_e4m3_sw(a) | ((unsigned int)f32_to_e4m3_sw(b) << 8);
#endif
}

__global__ __launch_bounds__(256) void quant_fp8_kernel(
    const float* __restrict__ x,
    const __hip_bfloat16* __restrict__ iscale,
    unsigned int* __restrict__ xq, int n4)
{
    int i = blockIdx.x * blockDim.x + threadIdx.x;
    if (i >= n4) return;
    const float inv = 1.0f / __bfloat162float(iscale[0]);
    float4 v = ((const float4*)x)[i];
    float a = fminf(fmaxf(v.x * inv, -448.0f), 448.0f);
    float b = fminf(fmaxf(v.y * inv, -448.0f), 448.0f);
    float c = fminf(fmaxf(v.z * inv, -448.0f), 448.0f);
    float d = fminf(fmaxf(v.w * inv, -448.0f), 448.0f);
    xq[i] = pack2_fp8(a, b) | (pack2_fp8(c, d) << 16);
}

__global__ __launch_bounds__(256) void fp8_gemm_kernel(
    const unsigned char* __restrict__ Aq,
    const unsigned char* __restrict__ Wq,
    const float* __restrict__ wscale,
    const __hip_bfloat16* __restrict__ iscale,
    float* __restrict__ out)
{
    __shared__ __align__(16) unsigned char lA[2][BM * BK];
    __shared__ __align__(16) unsigned char lB[2][BN * BK];

    const int tid  = threadIdx.x;
    const int lane = tid & 31;
    const int wave = tid >> 5;
    const int wm   = wave & 3;
    const int wn   = wave >> 2;

    const int bm = blockIdx.y * BM;
    const int bn = blockIdx.x * BN;

    const int lrow  = tid >> 1;
    const int lhalf = (tid & 1) * 64;

    const unsigned char* gA = Aq + (size_t)(bm + lrow) * K_TOTAL + lhalf;
    const unsigned char* gB = Wq + (size_t)(bn + lrow) * K_TOTAL + lhalf;
    const int ldsOff = lrow * BK + lhalf;

#if HAVE_ASYNC_LDS
#pragma unroll
    for (int i = 0; i < 4; ++i) {
        async_copy_b128(gA + i * 16, &lA[0][ldsOff + i * 16]);
        async_copy_b128(gB + i * 16, &lB[0][ldsOff + i * 16]);
    }
    wait_async0();
#else
    uint4 ra[4], rb[4];
#pragma unroll
    for (int i = 0; i < 4; ++i) {
        ra[i] = *(const uint4*)(gA + i * 16);
        rb[i] = *(const uint4*)(gB + i * 16);
    }
#pragma unroll
    for (int i = 0; i < 4; ++i) {
        *(uint4*)(&lA[0][ldsOff + i * 16]) = ra[i];
        *(uint4*)(&lB[0][ldsOff + i * 16]) = rb[i];
    }
#endif
    __syncthreads();

    v8f acc[2][4];
#pragma unroll
    for (int i = 0; i < 2; ++i)
#pragma unroll
        for (int j = 0; j < 4; ++j)
            acc[i][j] = (v8f){0.f, 0.f, 0.f, 0.f, 0.f, 0.f, 0.f, 0.f};

    const int l16 = lane & 15;
    const int hi  = lane >> 4;

    for (int ks = 0; ks < KSTAGES; ++ks) {
        const int cur = ks & 1;

        if (ks + 1 < KSTAGES) {
            const unsigned char* pa = gA + (size_t)(ks + 1) * BK;
            const unsigned char* pb = gB + (size_t)(ks + 1) * BK;
#if HAVE_ASYNC_LDS
#pragma unroll
            for (int i = 0; i < 4; ++i) {
                async_copy_b128(pa + i * 16, &lA[cur ^ 1][ldsOff + i * 16]);
                async_copy_b128(pb + i * 16, &lB[cur ^ 1][ldsOff + i * 16]);
            }
#else
#pragma unroll
            for (int i = 0; i < 4; ++i) {
                ra[i] = *(const uint4*)(pa + i * 16);
                rb[i] = *(const uint4*)(pb + i * 16);
            }
#endif
        }

        v16i afrag[2];
#pragma unroll
        for (int t = 0; t < 2; ++t) {
            const int row = wm * 32 + t * 16 + l16;
            const unsigned char* base = &lA[cur][row * BK + hi * 16];
            union { v16i v; uint2 u[8]; } fa;
#pragma unroll
            for (int i = 0; i < 8; ++i)
                fa.u[i] = *(const uint2*)(base + (i >> 1) * 32 + (i & 1) * 8);
            afrag[t] = fa.v;
        }

        v16i bfrag[4];
#pragma unroll
        for (int t = 0; t < 4; ++t) {
            const int col = wn * 64 + t * 16 + l16;
            const unsigned char* base = &lB[cur][col * BK + hi * 16];
            union { v16i v; uint4 u[4]; } fb;
#pragma unroll
            for (int i = 0; i < 4; ++i)
                fb.u[i] = *(const uint4*)(base + i * 32);
            bfrag[t] = fb.v;
        }

#pragma unroll
        for (int i = 0; i < 2; ++i)
#pragma unroll
            for (int j = 0; j < 4; ++j)
                acc[i][j] = __builtin_amdgcn_wmma_f32_16x16x128_fp8_fp8(
                    afrag[i], bfrag[j], (short)0, acc[i][j], false, false);

#if !HAVE_ASYNC_LDS
        if (ks + 1 < KSTAGES) {
#pragma unroll
            for (int i = 0; i < 4; ++i) {
                *(uint4*)(&lA[cur ^ 1][ldsOff + i * 16]) = ra[i];
                *(uint4*)(&lB[cur ^ 1][ldsOff + i * 16]) = rb[i];
            }
        }
#else
        if (ks + 1 < KSTAGES) wait_async0();
#endif
        __syncthreads();
    }

    const float scale = wscale[0] * __bfloat162float(iscale[0]);
#pragma unroll
    for (int i = 0; i < 2; ++i) {
        const int row0 = bm + wm * 32 + i * 16 + hi * 8;
#pragma unroll
        for (int j = 0; j < 4; ++j) {
            const int col = bn + wn * 64 + j * 16 + l16;
            float* o = out + (size_t)row0 * N_TOTAL + col;
#pragma unroll
            for (int v = 0; v < 8; ++v)
                o[(size_t)v * N_TOTAL] = acc[i][j][v] * scale;
        }
    }
}

extern "C" void kernel_launch(void* const* d_in, const int* in_sizes, int n_in,
                              void* d_out, int out_size, void* d_ws, size_t ws_size,
                              hipStream_t stream) {
    const float*           x      = (const float*)d_in[0];
    const unsigned char*   w      = (const unsigned char*)d_in[1];
    const float*           wscale = (const float*)d_in[2];
    const __hip_bfloat16*  iscale = (const __hip_bfloat16*)d_in[3];
    float*                 out    = (float*)d_out;
    unsigned int*          xq     = (unsigned int*)d_ws;

    const int n4 = (M_TOTAL * K_TOTAL) / 4;
    quant_fp8_kernel<<<n4 / 256, 256, 0, stream>>>(x, iscale, xq, n4);

    dim3 grid(N_TOTAL / BN, M_TOTAL / BM);
    fp8_gemm_kernel<<<grid, 256, 0, stream>>>(
        (const unsigned char*)xq, w, wscale, iscale, out);
}
